// ResNetBlock_18433999634782
// MI455X (gfx1250) — matrix operand order repaired and checked by reading, not compiled, not run on hardware
//
#include <hip/hip_runtime.h>
#include <hip/hip_bf16.h>

typedef __attribute__((ext_vector_type(16))) _Float16 v16h;
typedef __attribute__((ext_vector_type(8)))  _Float16 v8h;
typedef __attribute__((ext_vector_type(4)))  _Float16 v4h;
typedef __attribute__((ext_vector_type(8)))  float    v8f;

#define C_CH 27          // kernel offsets
#define NCH 64           // channels
#define EPSV 1e-5f
#define STAT_BLOCKS 640
#define ROWT 4           // 16-row tiles per wave (64 rows/wave)

// ---------------------------------------------------------------------------
// BN pass 1: per-channel partial sum / sumsq, fixed deterministic order.
// ---------------------------------------------------------------------------
__global__ void bn_stats_kernel(const float* __restrict__ x,
                                float* __restrict__ part, int n) {
    __shared__ float s_sum[256];
    __shared__ float s_sq[256];
    int t = threadIdx.x;
    int c = t & 63;
    int g = t >> 6;                    // 4 row-groups per block
    float sum = 0.f, sq = 0.f;
    for (int r = blockIdx.x * 4 + g; r < n; r += gridDim.x * 4) {
        float v = x[(size_t)r * NCH + c];
        sum += v;
        sq  += v * v;
    }
    s_sum[t] = sum;
    s_sq[t]  = sq;
    __syncthreads();
    if (t < 64) {
        float a = s_sum[t] + s_sum[t + 64] + s_sum[t + 128] + s_sum[t + 192];
        float b = s_sq[t]  + s_sq[t + 64]  + s_sq[t + 128]  + s_sq[t + 192];
        part[blockIdx.x * 128 + t]      = a;
        part[blockIdx.x * 128 + 64 + t] = b;
    }
}

// ---------------------------------------------------------------------------
// BN pass 2: reduce partials -> fused scale/shift.
// ---------------------------------------------------------------------------
__global__ void bn_reduce_kernel(const float* __restrict__ part,
                                 const float* __restrict__ gamma,
                                 const float* __restrict__ beta,
                                 float* __restrict__ ss, int nblocks, int n) {
    int c = threadIdx.x;
    if (c >= 64) return;
    float sum = 0.f, sq = 0.f;
    for (int b = 0; b < nblocks; ++b) {
        sum += part[b * 128 + c];
        sq  += part[b * 128 + 64 + c];
    }
    float invn  = 1.0f / (float)n;
    float mean  = sum * invn;
    float var   = sq * invn - mean * mean;
    float scale = gamma[c] * rsqrtf(var + EPSV);
    ss[c]      = scale;
    ss[64 + c] = beta[c] - mean * scale;
}

// ---------------------------------------------------------------------------
// BN pass 3: y = relu(x*scale+shift) -> f16 (4 elements/thread)
// ---------------------------------------------------------------------------
__global__ void bn_apply_kernel(const float* __restrict__ x,
                                const float* __restrict__ ss,
                                _Float16* __restrict__ y, int total4) {
    int i = blockIdx.x * blockDim.x + threadIdx.x;
    if (i >= total4) return;
    int base = i * 4;
    int c = base & 63;
    const float4 v = *(const float4*)(x + base);
    v4h o;
    o.x = (_Float16)fmaxf(v.x * ss[c + 0] + ss[64 + c + 0], 0.f);
    o.y = (_Float16)fmaxf(v.y * ss[c + 1] + ss[64 + c + 1], 0.f);
    o.z = (_Float16)fmaxf(v.z * ss[c + 2] + ss[64 + c + 2], 0.f);
    o.w = (_Float16)fmaxf(v.w * ss[c + 3] + ss[64 + c + 3], 0.f);
    *(v4h*)(y + base) = o;
}

// ---------------------------------------------------------------------------
// Pre-swizzle weights W[k][Cin][Cout] (f32) -> f16 WMMA B fragments.
// Fragment id = k*8 + kk*4 + nt ; per lane: column N = nt*16 + (lane&15),
// element h -> K = kk*32 + 8*(lane>>4) + (h&7) + 16*(h>>3).
// ---------------------------------------------------------------------------
__global__ void wswz_kernel(const float* __restrict__ W,
                            _Float16* __restrict__ out) {
    int tid = blockIdx.x * blockDim.x + threadIdx.x;
    if (tid >= C_CH * 8 * 32) return;
    int lane = tid & 31;
    int frag = tid >> 5;          // k*8 + kk*4 + nt
    int nt = frag & 3;
    int kk = (frag >> 2) & 1;
    int k  = frag >> 3;
    int kb = kk * 32 + (lane >> 4) * 8;
    const float* src = W + (size_t)k * NCH * NCH + nt * 16 + (lane & 15);
    _Float16* dst = out + (size_t)tid * 16;
#pragma unroll
    for (int h = 0; h < 16; ++h)
        dst[h] = (_Float16)src[(size_t)(kb + (h & 7) + (h >> 3) * 16) * NCH];
}

// ---------------------------------------------------------------------------
// Sparse conv via WMMA. Block = 256 threads (8 waves); wave = 64 rows x 64
// cols (4 row-tiles, register-blocked so each B fragment is reused 4x).
// Per offset k, the 8KB swizzled weight slice is staged into LDS by the
// whole block (register double-buffered), then read as ds_load B fragments.
// A fragments are gathered per the 16x32 f16 A layout:
//   lanes 0-15 : row M=lane,     K in {0..7, 16..23}  (+32*kk)
//   lanes 16-31: row M=lane-16,  K in {8..15, 24..31} (+32*kk)
// ---------------------------------------------------------------------------
__global__ __launch_bounds__(256)
void conv_wmma_kernel(const _Float16* __restrict__ y,
                      const int* __restrict__ nbr,
                      const _Float16* __restrict__ wswz,
                      const float* __restrict__ res,
                      float* __restrict__ out, int n) {
    __shared__ v16h lw[256];                 // one k-slice: 8 frags x 32 lanes
    const v16h* wz = (const v16h*)wswz;      // 256 v16h per offset k

    int t    = threadIdx.x;
    int lane = t & 31;
    int wid  = t >> 5;
    int row0 = (blockIdx.x * 8 + wid) * (16 * ROWT);
    bool active = row0 < n;                  // wave-uniform

    int rlocal = lane & 15;
    int half   = (lane >> 4) & 1;

    v8f acc[ROWT][4] = {};

    v16h wreg = wz[t];                       // k=0 slice element for this thread

    for (int k = 0; k < C_CH; ++k) {
        __syncthreads();                     // prior compute done reading lw
        lw[t] = wreg;
        __syncthreads();                     // slice k staged
        if (k + 1 < C_CH)
            wreg = wz[(size_t)(k + 1) * 256 + t];

        if (active) {
            int nrow[ROWT];
#pragma unroll
            for (int rt = 0; rt < ROWT; ++rt) {
                int r = row0 + rt * 16 + rlocal;
                if (r >= n) r = n - 1;       // per-lane select, EXEC unchanged
                nrow[rt] = nbr[(size_t)r * C_CH + k];
            }
#pragma unroll
            for (int kk = 0; kk < 2; ++kk) {
                int cb = kk * 32 + half * 8;
                v16h b[4];
#pragma unroll
                for (int nt = 0; nt < 4; ++nt)
                    b[nt] = lw[kk * 128 + nt * 32 + lane];
#pragma unroll
                for (int rt = 0; rt < ROWT; ++rt) {
                    const _Float16* yr = y + (size_t)nrow[rt] * NCH;
                    union { v16h v; v8h h[2]; } au;
                    au.h[0] = *(const v8h*)(yr + cb);       // K rel 0..7 (+8 hi)
                    au.h[1] = *(const v8h*)(yr + cb + 16);  // K rel 16..23 (+8 hi)
#pragma unroll
                    for (int nt = 0; nt < 4; ++nt)
                        acc[rt][nt] = __builtin_amdgcn_wmma_f32_16x16x32_f16(
                            false, au.v, false, b[nt], (short)0,
                            acc[rt][nt], false, false);
                }
            }
        }
    }

    if (!active) return;
    // D layout: lane -> col N = lane&15 (+16*nt); VGPR v -> row M = v + 8*half
    int colb = lane & 15;
#pragma unroll
    for (int rt = 0; rt < ROWT; ++rt) {
        int rbase = row0 + rt * 16 + half * 8;
#pragma unroll
        for (int v = 0; v < 8; ++v) {
            int r = rbase + v;
            if (r >= n) continue;
            size_t ro = (size_t)r * NCH;
            float r0 = acc[rt][0][v], r1 = acc[rt][1][v];
            float r2 = acc[rt][2][v], r3 = acc[rt][3][v];
            if (res) {
                r0 += res[ro + 0  + colb];
                r1 += res[ro + 16 + colb];
                r2 += res[ro + 32 + colb];
                r3 += res[ro + 48 + colb];
            }
            out[ro + 0  + colb] = r0;
            out[ro + 16 + colb] = r1;
            out[ro + 32 + colb] = r2;
            out[ro + 48 + colb] = r3;
        }
    }
}

// ---------------------------------------------------------------------------
static inline size_t align256(size_t x) { return (x + 255) & ~(size_t)255; }

extern "C" void kernel_launch(void* const* d_in, const int* in_sizes, int n_in,
                              void* d_out, int out_size, void* d_ws, size_t ws_size,
                              hipStream_t stream) {
    const float* x      = (const float*)d_in[0];
    const int*   nbr    = (const int*)d_in[1];   // int32 (JAX default x64-off)
    const float* W1     = (const float*)d_in[2];
    const float* gamma1 = (const float*)d_in[3];
    const float* beta1  = (const float*)d_in[4];
    const float* W2     = (const float*)d_in[5];
    const float* gamma2 = (const float*)d_in[6];
    const float* beta2  = (const float*)d_in[7];
    float* outp = (float*)d_out;

    const int n = in_sizes[0] / NCH;             // 200000 voxels

    // workspace carve-up
    char* ws = (char*)d_ws;
    size_t off = 0;
    _Float16* yh = (_Float16*)(ws + off); off = align256(off + (size_t)n * NCH * 2);
    float*    z  = (float*)   (ws + off); off = align256(off + (size_t)n * NCH * 4);
    _Float16* wz1 = (_Float16*)(ws + off); off = align256(off + (size_t)C_CH * 8 * 32 * 16 * 2);
    _Float16* wz2 = (_Float16*)(ws + off); off = align256(off + (size_t)C_CH * 8 * 32 * 16 * 2);
    float*    part = (float*)  (ws + off); off = align256(off + (size_t)STAT_BLOCKS * 128 * 4);
    float*    ss   = (float*)  (ws + off); off = align256(off + 128 * 4);
    (void)ws_size; (void)n_in; (void)out_size;

    const int total4     = n * NCH / 4;
    const int applyBlks  = (total4 + 255) / 256;
    const int waves      = (n + 16 * ROWT - 1) / (16 * ROWT);
    const int convBlks   = (waves + 7) / 8;           // 8 waves (256 thr)/block
    const int swzThreads = C_CH * 8 * 32;
    const int swzBlks    = (swzThreads + 255) / 256;

    // weight swizzle (independent of activations)
    wswz_kernel<<<swzBlks, 256, 0, stream>>>(W1, wz1);
    wswz_kernel<<<swzBlks, 256, 0, stream>>>(W2, wz2);

    // stage 1: BN(x) -> relu -> f16, conv1 -> z (f32)
    bn_stats_kernel<<<STAT_BLOCKS, 256, 0, stream>>>(x, part, n);
    bn_reduce_kernel<<<1, 64, 0, stream>>>(part, gamma1, beta1, ss, STAT_BLOCKS, n);
    bn_apply_kernel<<<applyBlks, 256, 0, stream>>>(x, ss, yh, total4);
    conv_wmma_kernel<<<convBlks, 256, 0, stream>>>(yh, nbr, wz1, nullptr, z, n);

    // stage 2: BN(z) -> relu -> f16 (reuse yh), conv2 + residual(x) -> out
    bn_stats_kernel<<<STAT_BLOCKS, 256, 0, stream>>>(z, part, n);
    bn_reduce_kernel<<<1, 64, 0, stream>>>(part, gamma2, beta2, ss, STAT_BLOCKS, n);
    bn_apply_kernel<<<applyBlks, 256, 0, stream>>>(z, ss, yh, total4);
    conv_wmma_kernel<<<convBlks, 256, 0, stream>>>(yh, nbr, wz2, x, outp, n);
}
